// MultiHeadAttention_50233937494513
// MI455X (gfx1250) — matrix operand order repaired and checked by reading, not compiled, not run on hardware
//
#include <hip/hip_runtime.h>
#include <hip/hip_bf16.h>

typedef __bf16 bf16;
typedef __attribute__((ext_vector_type(16))) __bf16 v16bf;
typedef __attribute__((ext_vector_type(8)))  __bf16 v8bf;
typedef __attribute__((ext_vector_type(4)))  __bf16 v4bf;
typedef __attribute__((ext_vector_type(8)))  float  v8f;

#define B_  4
#define S_  8192
#define D_  1024
#define H_  16
#define DH_ 64
#define SCALE_ 0.125f   // DH^-0.5
#define LDT 40          // LDS tile stride in bf16 (80B: 16B-aligned rows)

// ---------------- WMMA fragment helpers (CDNA5 16x16x32 bf16) ---------------
// A (16x32 MxK) row-major, per-lane = two contiguous 16B runs:
//   k in [half*8, half*8+8) and [16+half*8, 16+half*8+8)
__device__ __forceinline__ v16bf frag_a_k32(const bf16* base, int ld) {
  const int lane = threadIdx.x & 31;
  const int row  = lane & 15, half = lane >> 4;
  const v8bf lo = *(const v8bf*)(base + row * ld + half * 8);
  const v8bf hi = *(const v8bf*)(base + row * ld + 16 + half * 8);
  return __builtin_shufflevector(lo, hi, 0, 1, 2, 3, 4, 5, 6, 7,
                                         8, 9, 10, 11, 12, 13, 14, 15);
}
// B (32x16 KxN) staged TRANSPOSED as [N][K]: per-lane two 16B runs at
//   baseT[col*ld + half*8 .. +8) and baseT[col*ld + 16 + half*8 .. +8)
__device__ __forceinline__ v16bf frag_b_k32(const bf16* baseT, int ld) {
  const int lane = threadIdx.x & 31;
  const int col  = lane & 15, half = lane >> 4;
  const v8bf lo = *(const v8bf*)(baseT + col * ld + half * 8);
  const v8bf hi = *(const v8bf*)(baseT + col * ld + 16 + half * 8);
  return __builtin_shufflevector(lo, hi, 0, 1, 2, 3, 4, 5, 6, 7,
                                         8, 9, 10, 11, 12, 13, 14, 15);
}
__device__ __forceinline__ v8f wmma_bf16(const v16bf& a, const v16bf& b, const v8f& c) {
  return __builtin_amdgcn_wmma_f32_16x16x32_bf16(false, a, false, b, (short)0, c,
                                                 false, false);
}
__device__ __forceinline__ v4bf cvt4(const float4 f) {
  v4bf t; t[0] = (bf16)f.x; t[1] = (bf16)f.y; t[2] = (bf16)f.z; t[3] = (bf16)f.w;
  return t;
}
// CDNA5 async global->LDS 16B copy (ASYNCcnt domain), GV addressing mode.
__device__ __forceinline__ void async_copy_b128(void* lds_dst, const void* gsrc) {
  const unsigned lds_off = (unsigned)(uintptr_t)lds_dst;   // low 32 bits = LDS byte offset
  asm volatile("global_load_async_to_lds_b128 %0, %1, off"
               :: "v"(lds_off), "v"((unsigned long long)(uintptr_t)gsrc)
               : "memory");
}
__device__ __forceinline__ void wait_async0() {
  asm volatile("s_wait_asynccnt 0x0" ::: "memory");
}

// ---------------- K1: fused QKV GEMM  (x @ [Wq | Wkv]) -----------------------
// grid (256, 48), block 256. 128x64 tile per block, 16x64 per wave.
__global__ __launch_bounds__(256) void k_qkv_gemm(
    const float* __restrict__ x, const float* __restrict__ Wq,
    const float* __restrict__ Wkv,
    float* __restrict__ Qraw, float* __restrict__ Kraw, bf16* __restrict__ Vb)
{
  __shared__ __align__(16) bf16 As[128 * LDT];   // [row][k], 128x32
  __shared__ __align__(16) bf16 Bt[64 * LDT];    // [col][k], 64x32 (transposed)
  const int tid  = threadIdx.x;
  const int wave = tid >> 5;
  const int rowBase = blockIdx.x * 128;      // row in B*S
  const int colBase = blockIdx.y * 64;       // col in [0, 3072)

  const bool   isQ   = colBase < D_;
  const float* W     = isQ ? Wq : Wkv;
  const int    wld   = isQ ? D_ : 2 * D_;
  const int    wcol0 = isQ ? colBase : colBase - D_;

  v8f acc[4] = {};
  for (int k0 = 0; k0 < D_; k0 += 32) {
#pragma unroll
    for (int i = 0; i < 4; ++i) {            // x tile: 1024 float4 reads
      const int e = tid + i * 256;
      const int r = e >> 3, c4 = e & 7;
      const float4 f = *(const float4*)(x + (size_t)(rowBase + r) * D_ + k0 + c4 * 4);
      *(v4bf*)&As[r * LDT + c4 * 4] = cvt4(f);
    }
#pragma unroll
    for (int i = 0; i < 2; ++i) {            // W tile 32x64 -> transposed LDS
      const int e = tid + i * 256;
      const int r = e >> 4, c4 = e & 15;
      const float4 f = *(const float4*)(W + (size_t)(k0 + r) * wld + wcol0 + c4 * 4);
      const v4bf t = cvt4(f);
#pragma unroll
      for (int j = 0; j < 4; ++j) Bt[(c4 * 4 + j) * LDT + r] = t[j];
    }
    if (k0 + 32 < D_)                        // gfx1250 global_prefetch_b8
      __builtin_prefetch(&x[(size_t)(rowBase + wave * 16) * D_ + k0 + 32], 0, 1);
    __syncthreads();
    v16bf af = frag_a_k32(&As[(wave * 16) * LDT], LDT);
#pragma unroll
    for (int n = 0; n < 4; ++n) {
      v16bf bfrag = frag_b_k32(&Bt[(n * 16) * LDT], LDT);
      acc[n] = wmma_bf16(af, bfrag, acc[n]);
    }
    __syncthreads();
  }
  // Epilogue: scatter into (B,H,S,DH)
  const int lane = tid & 31;
  const int col = lane & 15, halfl = lane >> 4;
  const int r0 = rowBase + wave * 16;
  const int b  = r0 / S_;
  const int s0 = r0 % S_;
#pragma unroll
  for (int n = 0; n < 4; ++n) {
    const int gc = colBase + n * 16;
    if (gc < D_) {
      const int h = gc >> 6, dh = gc & 63;
      float* dst = Qraw + ((size_t)(b * H_ + h) * S_ + s0) * DH_ + dh;
#pragma unroll
      for (int v = 0; v < 8; ++v)
        dst[(size_t)(v + 8 * halfl) * DH_ + col] = acc[n][v];
    } else {
      const int c2 = gc - D_;
      const int h = c2 >> 7, j = c2 & 127;
      if (j < DH_) {
        float* dst = Kraw + ((size_t)(b * H_ + h) * S_ + s0) * DH_ + j;
#pragma unroll
        for (int v = 0; v < 8; ++v)
          dst[(size_t)(v + 8 * halfl) * DH_ + col] = acc[n][v];
      } else {
        bf16* dst = Vb + ((size_t)(b * H_ + h) * S_ + s0) * DH_ + (j - DH_);
#pragma unroll
        for (int v = 0; v < 8; ++v)
          dst[(size_t)(v + 8 * halfl) * DH_ + col] = (bf16)acc[n][v];
      }
    }
  }
}

// ---------------- K2: Q softmax over DH (row of 64), fused *SCALE -----------
__global__ __launch_bounds__(256) void k_q_softmax(
    const float* __restrict__ Qraw, bf16* __restrict__ Qs)
{
  const int wave = threadIdx.x >> 5;
  const int lane = threadIdx.x & 31;
  const size_t row = (size_t)blockIdx.x * 8 + wave;      // over B*H*S
  const float* src = Qraw + row * DH_;
  float v0 = src[lane], v1 = src[lane + 32];
  float m = fmaxf(v0, v1);
#pragma unroll
  for (int off = 16; off > 0; off >>= 1) m = fmaxf(m, __shfl_xor(m, off));
  float e0 = expf(v0 - m), e1 = expf(v1 - m);
  float s = e0 + e1;
#pragma unroll
  for (int off = 16; off > 0; off >>= 1) s += __shfl_xor(s, off);
  const float inv = SCALE_ / s;
  bf16* dst = Qs + row * DH_;
  dst[lane]      = (bf16)(e0 * inv);
  dst[lane + 32] = (bf16)(e1 * inv);
}

// ---------------- K3a: K column-softmax partial (max,sum) over s-chunks -----
__global__ __launch_bounds__(256) void k_k_stats(
    const float* __restrict__ Kraw, float* __restrict__ pmax,
    float* __restrict__ psum)
{
  __shared__ float lmx[256], lsm[256];
  const int bh = blockIdx.x, chunk = blockIdx.y;
  const int t = threadIdx.x;
  const int d = t & 63, sl = t >> 6;
  const float* base = Kraw + (size_t)bh * S_ * DH_ + d;
  float m = -3.0e38f, s = 0.f;
  for (int i = 0; i < 256; ++i) {
    const int srow = chunk * 1024 + sl + i * 4;
    const float xv = base[(size_t)srow * DH_];
    const float mn = fmaxf(m, xv);
    s = s * expf(m - mn) + expf(xv - mn);
    m = mn;
  }
  lmx[t] = m; lsm[t] = s;
  __syncthreads();
  if (t < 64) {
    float M = lmx[t];
    for (int q = 1; q < 4; ++q) M = fmaxf(M, lmx[t + 64 * q]);
    float SS = 0.f;
    for (int q = 0; q < 4; ++q) SS += lsm[t + 64 * q] * expf(lmx[t + 64 * q] - M);
    pmax[((size_t)bh * 8 + chunk) * 64 + t] = M;
    psum[((size_t)bh * 8 + chunk) * 64 + t] = SS;
  }
}

// ---------------- K3b: combine partials, normalize K -> bf16 -----------------
__global__ __launch_bounds__(256) void k_k_norm(
    const float* __restrict__ Kraw, const float* __restrict__ pmax,
    const float* __restrict__ psum, bf16* __restrict__ Ks)
{
  const int bh = blockIdx.x, chunk = blockIdx.y;
  const int t = threadIdx.x;
  const int d = t & 63, sl = t >> 6;
  float M = -3.0e38f;
  for (int q = 0; q < 8; ++q)
    M = fmaxf(M, pmax[((size_t)bh * 8 + q) * 64 + d]);
  float SS = 0.f;
  for (int q = 0; q < 8; ++q)
    SS += psum[((size_t)bh * 8 + q) * 64 + d] *
          expf(pmax[((size_t)bh * 8 + q) * 64 + d] - M);
  const float invS = 1.f / SS;
  const float* src = Kraw + (size_t)bh * S_ * DH_ + d;
  bf16*        dst = Ks   + (size_t)bh * S_ * DH_ + d;
  for (int i = 0; i < 256; ++i) {
    const int srow = chunk * 1024 + sl + i * 4;
    dst[(size_t)srow * DH_] = (bf16)(expf(src[(size_t)srow * DH_] - M) * invS);
  }
}

// ---------------- K4: ctx[d,e] = sum_s Ks[s,d] * V[s,e]  (per head) ----------
// grid 64 (bh). Both operands staged TRANSPOSED [d|e][s_local] so that the A
// fragment (= K^T tile) and B fragment are contiguous b128 loads.
__global__ __launch_bounds__(256) void k_ctx_gemm(
    const bf16* __restrict__ Ks, const bf16* __restrict__ Vb,
    bf16* __restrict__ ctx)
{
  __shared__ __align__(16) bf16 Kt[64 * LDT];   // [d][s_local]
  __shared__ __align__(16) bf16 Vt[64 * LDT];   // [e][s_local]
  const int bh = blockIdx.x;
  const int tid = threadIdx.x;
  const int wave = tid >> 5;
  const bf16* Kh = Ks + (size_t)bh * S_ * DH_;
  const bf16* Vh = Vb + (size_t)bh * S_ * DH_;
  v8f acc[2] = {};
  const int t0 = wave * 2;                    // 2 of 16 (d,e) tiles per wave
  const int dt = t0 >> 2;                     // same dt for both tiles
  const int sr = tid >> 3, sc8 = tid & 7;     // staging coords (32 s x 8 v8bf)
  for (int s0 = 0; s0 < S_; s0 += 32) {
    const v8bf kv = *(const v8bf*)(Kh + (size_t)(s0 + sr) * DH_ + sc8 * 8);
    const v8bf vv = *(const v8bf*)(Vh + (size_t)(s0 + sr) * DH_ + sc8 * 8);
#pragma unroll
    for (int j = 0; j < 8; ++j) {
      Kt[(sc8 * 8 + j) * LDT + sr] = kv[j];
      Vt[(sc8 * 8 + j) * LDT + sr] = vv[j];
    }
    __syncthreads();
    v16bf af = frag_a_k32(&Kt[(dt * 16) * LDT], LDT);
#pragma unroll
    for (int u = 0; u < 2; ++u) {
      const int et = (t0 + u) & 3;
      v16bf bfrag = frag_b_k32(&Vt[(et * 16) * LDT], LDT);
      acc[u] = wmma_bf16(af, bfrag, acc[u]);
    }
    __syncthreads();
  }
  const int lane = tid & 31;
  const int col = lane & 15, halfl = lane >> 4;
#pragma unroll
  for (int u = 0; u < 2; ++u) {
    const int et = (t0 + u) & 3;
    bf16* dst = ctx + (size_t)bh * DH_ * DH_ + (dt * 16) * DH_ + et * 16;
#pragma unroll
    for (int v = 0; v < 8; ++v)
      dst[(v + 8 * halfl) * DH_ + col] = (bf16)acc[u][v];
  }
}

// ---------------- K5: out2 = Qs @ ctx, write (B,S,H*DH) bf16 -----------------
// grid 4096. A fragments load straight from global (row-major, b128 pairs);
// ctx staged transposed once.
__global__ __launch_bounds__(256) void k_qctx_gemm(
    const bf16* __restrict__ Qs, const bf16* __restrict__ ctx,
    bf16* __restrict__ attn)
{
  __shared__ __align__(16) bf16 Ct[64 * LDT];   // [e][k]
  const int tid = threadIdx.x;
  const int wave = tid >> 5;
  const int rBase = blockIdx.x * 128;        // row in B*H*S
  const int bh = rBase >> 13;
  const bf16* ch = ctx + (size_t)bh * DH_ * DH_;
#pragma unroll
  for (int i = 0; i < 2; ++i) {              // 64x64 -> transposed LDS
    const int e = tid + i * 256;
    const int r = e >> 3, c8 = e & 7;
    const v8bf t = *(const v8bf*)(ch + r * DH_ + c8 * 8);
#pragma unroll
    for (int j = 0; j < 8; ++j) Ct[(c8 * 8 + j) * LDT + r] = t[j];
  }
  __syncthreads();
  const bf16* Arow = Qs + (size_t)(rBase + wave * 16) * DH_;
  v8f acc[4] = {};
#pragma unroll
  for (int k0 = 0; k0 < DH_; k0 += 32) {
    v16bf af = frag_a_k32(Arow + k0, DH_);   // global b128 pair per lane
#pragma unroll
    for (int n = 0; n < 4; ++n) {
      v16bf bfrag = frag_b_k32(&Ct[(n * 16) * LDT + k0], LDT);
      acc[n] = wmma_bf16(af, bfrag, acc[n]);
    }
  }
  const int lane = tid & 31;
  const int col = lane & 15, halfl = lane >> 4;
  const int r0 = rBase + wave * 16;
  const int b  = r0 >> 17;                   // / (H*S)
  const int h  = (r0 >> 13) & 15;
  const int s0 = r0 & 8191;
  bf16* dst = attn + ((size_t)b * S_ + s0) * D_ + h * DH_;
#pragma unroll
  for (int n = 0; n < 4; ++n)
#pragma unroll
    for (int v = 0; v < 8; ++v)
      dst[(size_t)(v + 8 * halfl) * D_ + n * 16 + col] = (bf16)acc[n][v];
}

// ---------------- K6: out = attn @ Wlin + blin  (fp32 out) -------------------
// A tile is a verbatim bf16 copy -> use CDNA5 async global->LDS DMA path.
__global__ __launch_bounds__(256) void k_out_gemm(
    const bf16* __restrict__ attn, const float* __restrict__ Wlin,
    const float* __restrict__ blin, float* __restrict__ out)
{
  __shared__ __align__(16) bf16 As[128 * LDT];
  __shared__ __align__(16) bf16 Bt[64 * LDT];
  const int tid = threadIdx.x;
  const int wave = tid >> 5;
  const int rowBase = blockIdx.x * 128;
  const int colBase = blockIdx.y * 64;
  v8f acc[4] = {};
  for (int k0 = 0; k0 < D_; k0 += 32) {
#pragma unroll
    for (int i = 0; i < 2; ++i) {            // attn tile: 512 async b128 copies
      const int e = tid + i * 256;
      const int r = e >> 2, c8 = e & 3;
      async_copy_b128(&As[r * LDT + c8 * 8],
                      attn + (size_t)(rowBase + r) * D_ + k0 + c8 * 8);
    }
#pragma unroll
    for (int i = 0; i < 2; ++i) {            // Wlin tile 32x64 -> transposed
      const int e = tid + i * 256;
      const int r = e >> 4, c4 = e & 15;
      const float4 f = *(const float4*)(Wlin + (size_t)(k0 + r) * D_ + colBase + c4 * 4);
      const v4bf t = cvt4(f);
#pragma unroll
      for (int j = 0; j < 4; ++j) Bt[(c4 * 4 + j) * LDT + r] = t[j];
    }
    wait_async0();                           // ASYNCcnt -> 0: A tile resident
    __syncthreads();
    v16bf af = frag_a_k32(&As[(wave * 16) * LDT], LDT);
#pragma unroll
    for (int n = 0; n < 4; ++n) {
      v16bf bfrag = frag_b_k32(&Bt[(n * 16) * LDT], LDT);
      acc[n] = wmma_bf16(af, bfrag, acc[n]);
    }
    __syncthreads();
  }
  const int lane = tid & 31;
  const int col = lane & 15, halfl = lane >> 4;
  const int r0 = rowBase + wave * 16;
#pragma unroll
  for (int n = 0; n < 4; ++n) {
    const float bias = blin[colBase + n * 16 + col];
#pragma unroll
    for (int v = 0; v < 8; ++v)
      out[(size_t)(r0 + v + 8 * halfl) * D_ + colBase + n * 16 + col] =
          acc[n][v] + bias;
  }
}

// ---------------- launch -----------------------------------------------------
extern "C" void kernel_launch(void* const* d_in, const int* in_sizes, int n_in,
                              void* d_out, int out_size, void* d_ws, size_t ws_size,
                              hipStream_t stream) {
  (void)in_sizes; (void)n_in; (void)out_size; (void)ws_size;
  const float* x    = (const float*)d_in[0];
  const float* Wq   = (const float*)d_in[1];
  const float* Wkv  = (const float*)d_in[2];
  const float* Wlin = (const float*)d_in[3];
  const float* blin = (const float*)d_in[4];
  float* out = (float*)d_out;

  char* ws = (char*)d_ws;
  size_t off = 0;
  auto carve = [&](size_t bytes) {
    void* p = ws + off; off += (bytes + 255) & ~(size_t)255; return p;
  };
  const size_t NE = (size_t)B_ * S_ * D_;            // 33,554,432 elements
  float* Qraw = (float*)carve(NE * 4);
  float* Kraw = (float*)carve(NE * 4);
  bf16*  Vb   = (bf16*) carve(NE * 2);
  bf16*  Qs   = (bf16*) carve(NE * 2);
  bf16*  Ks   = (bf16*) carve(NE * 2);
  bf16*  ctx  = (bf16*) carve((size_t)B_ * H_ * DH_ * DH_ * 2);
  float* pmax = (float*)carve((size_t)B_ * H_ * 8 * 64 * 4);
  float* psum = (float*)carve((size_t)B_ * H_ * 8 * 64 * 4);
  bf16*  attn = (bf16*)Kraw;       // alias: Kraw is dead after k_k_norm

  dim3 blk(256);
  k_qkv_gemm <<<dim3(256, 48), blk, 0, stream>>>(x, Wq, Wkv, Qraw, Kraw, Vb);
  k_q_softmax<<<dim3(65536),   blk, 0, stream>>>(Qraw, Qs);
  k_k_stats  <<<dim3(64, 8),   blk, 0, stream>>>(Kraw, pmax, psum);
  k_k_norm   <<<dim3(64, 8),   blk, 0, stream>>>(Kraw, pmax, psum, Ks);
  k_ctx_gemm <<<dim3(64),      blk, 0, stream>>>(Ks, Vb, ctx);
  k_qctx_gemm<<<dim3(4096),    blk, 0, stream>>>(Qs, ctx, attn);
  k_out_gemm <<<dim3(256, 16), blk, 0, stream>>>(attn, Wlin, blin, out);
}
